// CopyDecoder_33260226740801
// MI455X (gfx1250) — matrix operand order repaired and checked by reading, not compiled, not run on hardware
//
#include <hip/hip_runtime.h>
#include <hip/hip_bf16.h>
#include <cstdint>

// ---------------- problem constants ----------------
constexpr int B    = 32;
constexpr int SEQ  = 400;
constexpr int H    = 512;
constexpr int H2   = 1024;   // 2H
constexpr int E    = 256;
constexpr int V    = 32000;
constexpr int OOV  = 50;
constexpr int T    = 64;
constexpr int G3   = 1536;   // 3H
constexpr int XDIM = 1280;   // E + 2H
constexpr int WID  = V + SEQ;    // 32400 softmax width
constexpr int VO   = V + OOV;    // 32050 output width
constexpr int UNK  = 1;
constexpr float NEGV = -1000.0f;
constexpr float EPSV = 1e-05f;

// ---------------- WMMA types ----------------
typedef __attribute__((ext_vector_type(16))) __bf16 v16bf;
typedef __attribute__((ext_vector_type(8)))  float  v8f;

union AFrag { v16bf v; unsigned int u[8]; };
union BFrag { v16bf v; uint4 q[2]; };

__device__ __forceinline__ unsigned short f2bf(float f) {
  union { float f; uint32_t u; } x; x.f = f;
  uint32_t r = x.u + 0x7FFFu + ((x.u >> 16) & 1u);   // round-to-nearest-even
  return (unsigned short)(r >> 16);
}
__device__ __forceinline__ float sigmoidf(float x) { return 1.0f / (1.0f + expf(-x)); }

// ---------------------------------------------------------------------------
// TN bf16 GEMM, 32x16 macro-tile per wave (two 16x16 WMMA tiles sharing one
// B fragment -> halves B-matrix traffic; the vocab GEMM is B-bandwidth-bound).
// C[m,n] = act( sum_k A[m,k]*Bt[n,k] + bias[n] )
// A: M x K row-major bf16 (lda), Bt: N x K row-major bf16 (ldb), C fp32 (ldc).
// Requires: M % 32 == 0, N % 16 == 0, K % 32 == 0.
// Fragment layouts per CDNA5 ISA 7.12.2 (16-bit A 16x32 / B 32x16, wave32).
// ---------------------------------------------------------------------------
__global__ void gemm_tn_bf16(const unsigned short* __restrict__ A, int lda,
                             const unsigned short* __restrict__ Bt, int ldb,
                             const float* __restrict__ bias,
                             float* __restrict__ C, int ldc,
                             int M, int N, int K, int act) {
  int wid  = (blockIdx.x * blockDim.x + threadIdx.x) >> 5;   // wave id (uniform)
  int lane = threadIdx.x & 31;
  int ntn  = N >> 4;
  int ntiles = (M >> 5) * ntn;                               // 32-row macro tiles
  if (wid >= ntiles) return;                                 // wave-uniform exit
  int m0 = (wid / ntn) << 5;
  int n0 = (wid % ntn) << 4;

  int row  = lane & 15;
  int half = lane >> 4;
  const unsigned short* Arow0 = A  + (size_t)(m0 + row)      * lda + (half << 3); // +0/8
  const unsigned short* Arow1 = A  + (size_t)(m0 + 16 + row) * lda + (half << 3);
  const unsigned short* Brow  = Bt + (size_t)(n0 + row)      * ldb + (half << 3); // +0/8

  v8f acc0 = {};
  v8f acc1 = {};
  for (int k = 0; k < K; k += 32) {
    AFrag a0, a1; BFrag b;
    const unsigned short* ap0 = Arow0 + k;
    const unsigned short* ap1 = Arow1 + k;
#pragma unroll
    for (int i = 0; i < 4; ++i) a0.u[i]     = *(const unsigned int*)(ap0 + 2 * i);
#pragma unroll
    for (int i = 0; i < 4; ++i) a0.u[4 + i] = *(const unsigned int*)(ap0 + 16 + 2 * i);
#pragma unroll
    for (int i = 0; i < 4; ++i) a1.u[i]     = *(const unsigned int*)(ap1 + 2 * i);
#pragma unroll
    for (int i = 0; i < 4; ++i) a1.u[4 + i] = *(const unsigned int*)(ap1 + 16 + 2 * i);
    b.q[0] = *(const uint4*)(Brow + k);
    b.q[1] = *(const uint4*)(Brow + k + 16);
    acc0 = __builtin_amdgcn_wmma_f32_16x16x32_bf16(false, a0.v, false, b.v,
                                                   (short)0, acc0, false, false);
    acc1 = __builtin_amdgcn_wmma_f32_16x16x32_bf16(false, a1.v, false, b.v,
                                                   (short)0, acc1, false, false);
  }

  int n = n0 + (lane & 15);
  float bv = bias ? bias[n] : 0.0f;
#pragma unroll
  for (int r = 0; r < 8; ++r) {
    int m = m0 + ((lane >> 4) << 3) + r;
    float v0 = acc0[r] + bv;
    float v1 = acc1[r] + bv;
    if (act == 1) { v0 = tanhf(v0); v1 = tanhf(v1); }
    C[(size_t)m * ldc + n]        = v0;
    C[(size_t)(m + 16) * ldc + n] = v1;
  }
}

// ---------------- elementwise fp32 -> bf16 ----------------
__global__ void convert_bf16_k(const float* __restrict__ src,
                               unsigned short* __restrict__ dst, long n) {
  long i = (long)blockIdx.x * blockDim.x + threadIdx.x;
  long stride = (long)gridDim.x * blockDim.x;
  for (; i < n; i += stride) dst[i] = f2bf(src[i]);
}

__global__ void zero_f32_k(float* __restrict__ p, int n) {
  int i = blockIdx.x * blockDim.x + threadIdx.x;
  if (i < n) p[i] = 0.0f;
}

// ---------------- LDS-tiled transpose: src fp32 (R x C) -> dst bf16 (C x R) ---
__global__ void transpose_f32_bf16_k(const float* __restrict__ src,
                                     unsigned short* __restrict__ dst,
                                     int R, int C) {
  __shared__ float tile[32][33];
  int tx = threadIdx.x & 31;
  int ty = threadIdx.x >> 5;       // 0..7 (blockDim = 256)
  int c0 = blockIdx.x * 32;
  int r0 = blockIdx.y * 32;
#pragma unroll
  for (int i = 0; i < 32; i += 8) {
    int r = r0 + ty + i, c = c0 + tx;
    tile[ty + i][tx] = (r < R && c < C) ? src[(size_t)r * C + c] : 0.0f;
  }
  __syncthreads();
#pragma unroll
  for (int i = 0; i < 32; i += 8) {
    int c = c0 + ty + i, r = r0 + tx;
    if (c < C && r < R) dst[(size_t)c * R + r] = f2bf(tile[tx][ty + i]);
  }
}

// ---------------- src_lens + gather last encoder state (bf16) ----------------
__global__ void gather_last_k(const float* __restrict__ enc,
                              const int* __restrict__ sources,
                              unsigned short* __restrict__ last_bf) {
  __shared__ int red[256];
  __shared__ int s_len;
  int b = blockIdx.x, tid = threadIdx.x;
  int cnt = 0;
  for (int s = tid; s < SEQ; s += 256) cnt += (sources[b * SEQ + s] > 0) ? 1 : 0;
  red[tid] = cnt;
  __syncthreads();
  for (int off = 128; off; off >>= 1) {
    if (tid < off) red[tid] += red[tid + off];
    __syncthreads();
  }
  if (tid == 0) s_len = red[0];
  __syncthreads();
  int last = s_len - 1;
  const float* row = enc + ((size_t)b * SEQ + last) * H2;
  for (int k = tid; k < H2; k += 256) last_bf[b * H2 + k] = f2bf(row[k]);
}

// ---------------- build x = [embedding[tok], weighted] in bf16 ----------------
__global__ void prep_x_k(const float* __restrict__ emb,
                         const float* __restrict__ weighted,
                         const int* __restrict__ targets,
                         unsigned short* __restrict__ x_bf, int t) {
  int i = blockIdx.x * blockDim.x + threadIdx.x;
  if (i >= B * XDIM) return;
  int b = i / XDIM, j = i - b * XDIM;
  float v;
  if (j < E) {
    int tok = targets[b * T + t];
    if (tok >= V) tok = UNK;
    v = emb[(size_t)tok * E + j];
  } else {
    v = weighted[b * H2 + (j - E)];
  }
  x_bf[i] = f2bf(v);
}

// ---------------- GRU gate fuse ----------------
__global__ void gru_gate_k(const float* __restrict__ gi,
                           const float* __restrict__ gh,
                           float* __restrict__ state,
                           unsigned short* __restrict__ state_bf) {
  int i = blockIdx.x * blockDim.x + threadIdx.x;
  if (i >= B * H) return;
  int b = i / H, j = i - b * H;
  const float* gib = gi + b * G3;
  const float* ghb = gh + b * G3;
  float r = sigmoidf(gib[j] + ghb[j]);
  float z = sigmoidf(gib[j + H] + ghb[j + H]);
  float n = tanhf(gib[j + 2 * H] + r * ghb[j + 2 * H]);
  float s = (1.0f - z) * n + z * state[i];
  state[i] = s;
  state_bf[i] = f2bf(s);
}

// ---------------- attention scores: one wave per (b,s) ----------------
__global__ void attn_scores_k(const float* __restrict__ enc_proj,
                              const float* __restrict__ state,
                              const int* __restrict__ sources,
                              float* __restrict__ scores) {
  int gid  = blockIdx.x * blockDim.x + threadIdx.x;
  int w    = gid >> 5;
  int lane = gid & 31;
  if (w >= B * SEQ) return;
  int b = w / SEQ, s = w - b * SEQ;
  const float* ep = enc_proj + ((size_t)b * SEQ + s) * H;
  const float* st = state + b * H;
  float acc = 0.0f;
  for (int k = lane; k < H; k += 32) acc += ep[k] * st[k];
#pragma unroll
  for (int off = 16; off; off >>= 1) acc += __shfl_xor(acc, off, 32);
  if (lane == 0) {
    float sc = tanhf(acc) + ((sources[b * SEQ + s] == 0) ? NEGV : 0.0f);
    scores[(size_t)b * WID + V + s] = sc;
  }
}

// ---------------- softmax + copy-dist scatter + weighted context --------------
__global__ void __launch_bounds__(1024)
softmax_copy_k(const float* __restrict__ scores,
               const int* __restrict__ sources,
               const int* __restrict__ targets,
               const float* __restrict__ enc,
               float* __restrict__ out,
               float* __restrict__ weighted, int t) {
  __shared__ float red[1024];
  __shared__ float pc[SEQ];
  __shared__ float s_max, s_inv;
  __shared__ int   s_cnt;
  int b = blockIdx.x, tid = threadIdx.x;
  const float* sc = scores + (size_t)b * WID;

  // 1) max
  float m = -INFINITY;
  for (int i = tid; i < WID; i += 1024) m = fmaxf(m, sc[i]);
  red[tid] = m; __syncthreads();
  for (int off = 512; off; off >>= 1) {
    if (tid < off) red[tid] = fmaxf(red[tid], red[tid + off]);
    __syncthreads();
  }
  if (tid == 0) s_max = red[0];
  __syncthreads();
  float mx = s_max;

  // 2) sum(exp)
  float sum = 0.0f;
  for (int i = tid; i < WID; i += 1024) sum += expf(sc[i] - mx);
  red[tid] = sum; __syncthreads();
  for (int off = 512; off; off >>= 1) {
    if (tid < off) red[tid] += red[tid + off];
    __syncthreads();
  }
  if (tid == 0) s_inv = 1.0f / red[0];
  __syncthreads();
  float inv = s_inv;

  // 3) base output row: prob_g for v<V, EPS for OOV slots
  float* orow = out + ((size_t)b * T + t) * VO;
  for (int i = tid; i < V; i += 1024) orow[i] = expf(sc[i] - mx) * inv;
  for (int i = V + tid; i < VO; i += 1024) orow[i] = EPSV;
  if (tid == 0) s_cnt = 0;
  __threadfence();
  __syncthreads();

  // 4) copy probs: scatter-add into vocab, count target matches
  int tgt = targets[b * T + t];
  int cnt = 0;
  for (int s = tid; s < SEQ; s += 1024) {
    float p = expf(sc[V + s] - mx) * inv;
    pc[s] = p;
    int src = sources[b * SEQ + s];
    atomicAdd(&orow[src], p);
    if (src == tgt) cnt++;
  }
  if (cnt) atomicAdd(&s_cnt, cnt);
  __syncthreads();

  // 5) selection weights for next context
  float scale = (s_cnt > 1) ? (1.0f / (float)s_cnt) : 1.0f;
  for (int s = tid; s < SEQ; s += 1024)
    pc[s] = (sources[b * SEQ + s] == tgt) ? pc[s] * scale : 0.0f;
  __syncthreads();

  // 6) weighted[b,h] = sum_s pc[s] * enc[b,s,h]
  for (int h = tid; h < H2; h += 1024) {
    float acc = 0.0f;
    const float* eb = enc + (size_t)b * SEQ * H2 + h;
    for (int s = 0; s < SEQ; ++s) acc += pc[s] * eb[(size_t)s * H2];
    weighted[b * H2 + h] = acc;
  }
}

// ---------------------------------------------------------------------------
extern "C" void kernel_launch(void* const* d_in, const int* in_sizes, int n_in,
                              void* d_out, int out_size, void* d_ws, size_t ws_size,
                              hipStream_t stream) {
  const float* enc   = (const float*)d_in[0];
  const int*   srcs  = (const int*)d_in[1];
  const int*   tgts  = (const int*)d_in[2];
  const float* emb   = (const float*)d_in[3];
  const float* Wih   = (const float*)d_in[4];
  const float* Whh   = (const float*)d_in[5];
  const float* bih   = (const float*)d_in[6];
  const float* bhh   = (const float*)d_in[7];
  const float* Ws_w  = (const float*)d_in[8];
  const float* Ws_b  = (const float*)d_in[9];
  const float* Wc_w  = (const float*)d_in[10];
  const float* Wc_b  = (const float*)d_in[11];
  const float* Wo_w  = (const float*)d_in[12];
  const float* Wo_b  = (const float*)d_in[13];
  float* out = (float*)d_out;

  // workspace carve-up (256B aligned)
  size_t off = 0;
  auto carve = [&](size_t bytes) {
    void* p = (char*)d_ws + off;
    off += (bytes + 255) & ~(size_t)255;
    return p;
  };
  unsigned short* WoT_bf = (unsigned short*)carve((size_t)V * H * 2);      // 31.3 MB
  unsigned short* Wih_bf = (unsigned short*)carve((size_t)G3 * XDIM * 2);  // 3.75 MB
  unsigned short* Whh_bf = (unsigned short*)carve((size_t)G3 * H * 2);
  unsigned short* WcT_bf = (unsigned short*)carve((size_t)H * H2 * 2);
  unsigned short* WsT_bf = (unsigned short*)carve((size_t)H * H2 * 2);
  unsigned short* enc_bf = (unsigned short*)carve((size_t)B * SEQ * H2 * 2); // 25 MB
  float* enc_proj        = (float*)carve((size_t)B * SEQ * H * 4);          // 25 MB
  unsigned short* last_bf  = (unsigned short*)carve((size_t)B * H2 * 2);
  float* state             = (float*)carve((size_t)B * H * 4);
  unsigned short* state_bf = (unsigned short*)carve((size_t)B * H * 2);
  float* weighted          = (float*)carve((size_t)B * H2 * 4);
  unsigned short* x_bf     = (unsigned short*)carve((size_t)B * XDIM * 2);
  float* gi                = (float*)carve((size_t)B * G3 * 4);
  float* gh                = (float*)carve((size_t)B * G3 * 4);
  float* scores            = (float*)carve((size_t)B * WID * 4);            // 4 MB

  // waves = (M/32)*(N/16), 8 waves per 256-thread block
  auto gemm_blocks = [](int M, int N) { return ((M / 32) * (N / 16) + 7) / 8; };

  // ---- one-time per call: weight conversions / transposes ----
  convert_bf16_k<<<1024, 256, 0, stream>>>(Wih, Wih_bf, (long)G3 * XDIM);
  convert_bf16_k<<<1024, 256, 0, stream>>>(Whh, Whh_bf, (long)G3 * H);
  convert_bf16_k<<<2048, 256, 0, stream>>>(enc, enc_bf, (long)B * SEQ * H2);
  transpose_f32_bf16_k<<<dim3(V / 32, H / 32), 256, 0, stream>>>(Wo_w, WoT_bf, H, V);
  transpose_f32_bf16_k<<<dim3(H / 32, H2 / 32), 256, 0, stream>>>(Wc_w, WcT_bf, H2, H);
  transpose_f32_bf16_k<<<dim3(H / 32, H2 / 32), 256, 0, stream>>>(Ws_w, WsT_bf, H2, H);
  zero_f32_k<<<(B * H2 + 255) / 256, 256, 0, stream>>>(weighted, B * H2);
  gather_last_k<<<B, 256, 0, stream>>>(enc, srcs, last_bf);

  // state0 = last @ Ws_w + Ws_b   (M=32, N=512, K=1024)
  gemm_tn_bf16<<<gemm_blocks(B, H), 256, 0, stream>>>(
      last_bf, H2, WsT_bf, H2, Ws_b, state, H, B, H, H2, 0);
  convert_bf16_k<<<64, 256, 0, stream>>>(state, state_bf, (long)B * H);

  // enc_proj = tanh(enc @ Wc_w + Wc_b)   (M=12800, N=512, K=1024)
  gemm_tn_bf16<<<gemm_blocks(B * SEQ, H), 256, 0, stream>>>(
      enc_bf, H2, WcT_bf, H2, Wc_b, enc_proj, H, B * SEQ, H, H2, 1);

  // ---- sequential decode loop ----
  for (int t = 0; t < T; ++t) {
    prep_x_k<<<(B * XDIM + 255) / 256, 256, 0, stream>>>(emb, weighted, tgts, x_bf, t);

    // gi = x @ Wih^T + bih  (M=32, N=1536, K=1280)
    gemm_tn_bf16<<<gemm_blocks(B, G3), 256, 0, stream>>>(
        x_bf, XDIM, Wih_bf, XDIM, bih, gi, G3, B, G3, XDIM, 0);
    // gh = state @ Whh^T + bhh  (M=32, N=1536, K=512)
    gemm_tn_bf16<<<gemm_blocks(B, G3), 256, 0, stream>>>(
        state_bf, H, Whh_bf, H, bhh, gh, G3, B, G3, H, 0);

    gru_gate_k<<<(B * H + 255) / 256, 256, 0, stream>>>(gi, gh, state, state_bf);

    // score_g = state @ Wo_w + Wo_b  (M=32, N=32000, K=512) -> scores[:, :V]
    gemm_tn_bf16<<<gemm_blocks(B, V), 256, 0, stream>>>(
        state_bf, H, WoT_bf, H, Wo_b, scores, WID, B, V, H, 0);

    attn_scores_k<<<(B * SEQ * 32 + 255) / 256, 256, 0, stream>>>(
        enc_proj, state, srcs, scores);

    softmax_copy_k<<<B, 1024, 0, stream>>>(scores, srcs, tgts, enc, out, weighted, t);
  }
  (void)in_sizes; (void)n_in; (void)out_size; (void)ws_size;
}
